// Attention_62337155334878
// MI455X (gfx1250) — matrix operand order repaired and checked by reading, not compiled, not run on hardware
//
#include <hip/hip_runtime.h>

// ---------------------------------------------------------------------------
// MHA forward for MI455X (gfx1250), bf16 WMMA (16x16x32) with f32 accumulate.
// Pipeline: cvt(f32->bf16) -> QKV GEMM -> flash-attention -> OUT GEMM(+bias).
// Workspace use: ~48 MB (whole working set fits in the 192 MB L2).
// ---------------------------------------------------------------------------

typedef __bf16 bf16;
typedef __attribute__((ext_vector_type(16))) __bf16 v16bf;
typedef __attribute__((ext_vector_type(8)))  __bf16 v8bf;
typedef __attribute__((ext_vector_type(4)))  __bf16 v4bf;
typedef __attribute__((ext_vector_type(8)))  float  v8f;

#define B_   2
#define N_   2048
#define DIM_ 1024
#define H_   16
#define D_   64

__device__ __forceinline__ bf16 f2bf(float f) { return (bf16)f; }

__device__ __forceinline__ v16bf make_frag(v8bf lo, v8bf hi) {
  v16bf r;
#pragma unroll
  for (int i = 0; i < 8; ++i) { r[i] = lo[i]; r[i + 8] = hi[i]; }
  return r;
}

__device__ __forceinline__ v8f wmma_bf16(v16bf a, v16bf b, v8f c) {
  // (neg_a, A, neg_b, B, c_mod, C, reuse_a, reuse_b)
  return __builtin_amdgcn_wmma_f32_16x16x32_bf16(false, a, false, b,
                                                 (short)0, c, false, false);
}

// A fragment: 16(M) x 32(K) bf16 tile of a row-major matrix (leading dim ld).
// Lane holds row (lane&15); ISA layout: lanes 0-15 get K {0..7,16..23},
// lanes 16-31 get K {8..15,24..31}  (two 16-byte chunks per lane).
__device__ __forceinline__ v16bf load_a_frag(const bf16* p, int ld, int row0,
                                             int kofs, int lane) {
  const bf16* r = p + (size_t)(row0 + (lane & 15)) * ld + kofs + ((lane & 16) ? 8 : 0);
  return make_frag(*(const v8bf*)r, *(const v8bf*)(r + 16));
}

// B fragment: 32(K) x 16(N) where B[k][n] = M[col0+n][kofs+k] for row-major M
// (i.e. B is a transposed slice of M). Lane holds column n = lane&15;
// lanes 0-15 get K {0..7,16..23}, lanes 16-31 get K {8..15,24..31} (two 16-byte chunks).
__device__ __forceinline__ v16bf load_bT_frag(const bf16* p, int ld, int col0,
                                              int kofs, int lane) {
  const bf16* r = p + (size_t)(col0 + (lane & 15)) * ld + kofs + ((lane & 16) ? 8 : 0);
  return make_frag(*(const v8bf*)r, *(const v8bf*)(r + 16));
}

// ---------------------------------------------------------------------------
// Stage 1: fp32 -> bf16 (vectorized x4)
// ---------------------------------------------------------------------------
__global__ __launch_bounds__(256)
void cvt_f32_bf16(const float* __restrict__ in, bf16* __restrict__ out, int n4) {
  const int i = blockIdx.x * 256 + threadIdx.x;
  if (i >= n4) return;
  const float4 v = ((const float4*)in)[i];
  v4bf o;
  o[0] = f2bf(v.x); o[1] = f2bf(v.y); o[2] = f2bf(v.z); o[3] = f2bf(v.w);
  ((v4bf*)out)[i] = o;
}

// ---------------------------------------------------------------------------
// Stage 2: QKV GEMM  C[4096,3072] = X[4096,1024] @ Wqkv^T
// Wave tile 64x64 (16 accum fragments); block = 8 waves = 128x256.
// Epilogue scatters Q,K as [bh, n, 64] and V transposed as [bh, 64, n].
// ---------------------------------------------------------------------------
__global__ __launch_bounds__(256)
void qkv_gemm(const bf16* __restrict__ X, const bf16* __restrict__ W,
              bf16* __restrict__ Qb, bf16* __restrict__ Kb, bf16* __restrict__ Vt) {
  const int lane = threadIdx.x & 31, wave = threadIdx.x >> 5;
  const int m0 = blockIdx.x * 128 + (wave >> 2) * 64;
  const int n0 = blockIdx.y * 256 + (wave & 3) * 64;

  const v8f zero = {};
  v8f acc[4][4];
#pragma unroll
  for (int i = 0; i < 4; ++i)
#pragma unroll
    for (int j = 0; j < 4; ++j) acc[i][j] = zero;

  for (int k = 0; k < DIM_; k += 32) {
    v16bf a[4], bb[4];
#pragma unroll
    for (int i = 0; i < 4; ++i) a[i] = load_a_frag(X, DIM_, m0 + 16 * i, k, lane);
#pragma unroll
    for (int j = 0; j < 4; ++j) bb[j] = load_bT_frag(W, DIM_, n0 + 16 * j, k, lane);
#pragma unroll
    for (int i = 0; i < 4; ++i)
#pragma unroll
      for (int j = 0; j < 4; ++j) acc[i][j] = wmma_bf16(a[i], bb[j], acc[i][j]);
  }

  // C/D layout: lane = N column, VGPR r -> M = r + 8*(lane>=16)
  const int nn = lane & 15, mo = (lane & 16) ? 8 : 0;
#pragma unroll
  for (int i = 0; i < 4; ++i) {
#pragma unroll
    for (int j = 0; j < 4; ++j) {
      const int f = n0 + 16 * j + nn;         // output feature 0..3071
      const int sel = f >> 10;                // 0=Q 1=K 2=V
      const int rem = f & 1023, h = rem >> 6, dc = rem & 63;
#pragma unroll
      for (int r = 0; r < 8; ++r) {
        const int m = m0 + 16 * i + r + mo;   // global row 0..4095
        const int b = m >> 11, tok = m & 2047;
        const bf16 v = f2bf(acc[i][j][r]);
        const size_t bh = (size_t)(b * H_ + h);
        if (sel == 0)      Qb[(bh * N_ + tok) * D_ + dc] = v;
        else if (sel == 1) Kb[(bh * N_ + tok) * D_ + dc] = v;
        else               Vt[(bh * D_ + dc) * N_ + tok] = v;  // V transposed
      }
    }
  }
}

// ---------------------------------------------------------------------------
// Stage 3: flash attention. One wave = 16 queries of one (b,h).
// Block = 4 waves (64 queries). scale = HEADS^-0.5 = 0.25 (faithful to ref).
// Key mask: key 0 always valid, key j>0 valid iff mask[b][j-1].
// ---------------------------------------------------------------------------
__global__ __launch_bounds__(128)
void attn_kernel(const bf16* __restrict__ Qb, const bf16* __restrict__ Kb,
                 const bf16* __restrict__ Vt, const unsigned char* __restrict__ mask,
                 bf16* __restrict__ Ab) {
  __shared__ __align__(16) bf16 lds_p[4][16 * 32];  // per-wave P staging tile

  const int lane = threadIdx.x & 31, wave = threadIdx.x >> 5;
  const int bh = blockIdx.x;                 // 0..31
  const int b = bh >> 4, h = bh & 15;
  const int q0 = (blockIdx.y * 4 + wave) * 16;

  const bf16* Qh = Qb + (size_t)bh * N_ * D_;
  const bf16* Kh = Kb + (size_t)bh * N_ * D_;
  const bf16* Vh = Vt + (size_t)bh * D_ * N_;
  const unsigned char* mb = mask + (size_t)b * (N_ - 1);

  // Q fragments for the whole loop (d=64 -> two K=32 chunks)
  const v16bf aq0 = load_a_frag(Qh, D_, q0, 0, lane);
  const v16bf aq1 = load_a_frag(Qh, D_, q0, 32, lane);

  const v8f zero = {};
  v8f o[4];
#pragma unroll
  for (int f = 0; f < 4; ++f) o[f] = zero;
  float mrow[8], lrow[8];
#pragma unroll
  for (int r = 0; r < 8; ++r) { mrow[r] = -3.0e38f; lrow[r] = 0.0f; }

  const int nn = lane & 15, mo = (lane & 16) ? 8 : 0;
  const int k0g = /*tile0 key col*/ nn, k1g = 16 + nn;

  for (int j = 0; j < N_; j += 32) {
    // S = Q @ K^T  (two 16x16 tiles: keys j..j+15 and j+16..j+31)
    v8f s0 = zero, s1 = zero;
    s0 = wmma_bf16(aq0, load_bT_frag(Kh, D_, j, 0, lane), s0);
    s0 = wmma_bf16(aq1, load_bT_frag(Kh, D_, j, 32, lane), s0);
    s1 = wmma_bf16(aq0, load_bT_frag(Kh, D_, j + 16, 0, lane), s1);
    s1 = wmma_bf16(aq1, load_bT_frag(Kh, D_, j + 16, 32, lane), s1);

    // key-side mask (additive -inf bias), applied per lane (= key column)
    const int kk0 = j + k0g, kk1 = j + k1g;
    const float bias0 = (kk0 == 0 || mb[kk0 - 1]) ? 0.0f : -3.0e38f;
    const float bias1 = (kk1 == 0 || mb[kk1 - 1]) ? 0.0f : -3.0e38f;

    // online softmax per row (M lives on VGPR index + lane-half; N on lanes)
#pragma unroll
    for (int r = 0; r < 8; ++r) {
      float e0 = s0[r] * 0.25f + bias0;
      float e1 = s1[r] * 0.25f + bias1;
      float v = fmaxf(e0, e1);
#pragma unroll
      for (int off = 1; off < 16; off <<= 1) v = fmaxf(v, __shfl_xor(v, off, 32));
      const float mn = fmaxf(mrow[r], v);
      const float alpha = __expf(mrow[r] - mn);
      const float p0 = __expf(e0 - mn);
      const float p1 = __expf(e1 - mn);
      float rs = p0 + p1;
#pragma unroll
      for (int off = 1; off < 16; off <<= 1) rs += __shfl_xor(rs, off, 32);
      lrow[r] = lrow[r] * alpha + rs;
      mrow[r] = mn;
#pragma unroll
      for (int f = 0; f < 4; ++f) o[f][r] *= alpha;

      // stash P (D layout) into LDS as a row-major 16x32 bf16 tile
      const int m = r + mo;
      lds_p[wave][m * 32 + nn]      = f2bf(p0);
      lds_p[wave][m * 32 + 16 + nn] = f2bf(p1);
    }

    asm volatile("s_wait_dscnt 0" ::: "memory");  // LDS stores visible to loads

    // reload P as a 16x32 A fragment, then O += P @ V (4 B fragments, d=64)
    {
      const bf16* base = &lds_p[wave][(lane & 15) * 32 + ((lane & 16) ? 8 : 0)];
      const v16bf pa = make_frag(*(const v8bf*)base, *(const v8bf*)(base + 16));
#pragma unroll
      for (int f = 0; f < 4; ++f)
        o[f] = wmma_bf16(pa, load_bT_frag(Vh, N_, f * 16, j, lane), o[f]);
    }
  }

  // epilogue: O /= l, write bf16 into attention-output [b, n, h*64 + d]
#pragma unroll
  for (int f = 0; f < 4; ++f) {
#pragma unroll
    for (int r = 0; r < 8; ++r) {
      const int tok = q0 + r + mo;
      Ab[((size_t)b * N_ + tok) * DIM_ + h * D_ + f * 16 + nn] =
          f2bf(o[f][r] / lrow[r]);
    }
  }
}

// ---------------------------------------------------------------------------
// Stage 4: OUT GEMM  out[4096,1024] = Ab @ Wout^T + b_out  (fp32 stores)
// ---------------------------------------------------------------------------
__global__ __launch_bounds__(256)
void out_gemm(const bf16* __restrict__ A, const bf16* __restrict__ W,
              const float* __restrict__ bias, float* __restrict__ out) {
  const int lane = threadIdx.x & 31, wave = threadIdx.x >> 5;
  const int m0 = blockIdx.x * 128 + (wave >> 2) * 64;
  const int n0 = blockIdx.y * 256 + (wave & 3) * 64;

  const v8f zero = {};
  v8f acc[4][4];
#pragma unroll
  for (int i = 0; i < 4; ++i)
#pragma unroll
    for (int j = 0; j < 4; ++j) acc[i][j] = zero;

  for (int k = 0; k < DIM_; k += 32) {
    v16bf a[4], bb[4];
#pragma unroll
    for (int i = 0; i < 4; ++i) a[i] = load_a_frag(A, DIM_, m0 + 16 * i, k, lane);
#pragma unroll
    for (int j = 0; j < 4; ++j) bb[j] = load_bT_frag(W, DIM_, n0 + 16 * j, k, lane);
#pragma unroll
    for (int i = 0; i < 4; ++i)
#pragma unroll
      for (int j = 0; j < 4; ++j) acc[i][j] = wmma_bf16(a[i], bb[j], acc[i][j]);
  }

  const int nn = lane & 15, mo = (lane & 16) ? 8 : 0;
#pragma unroll
  for (int j = 0; j < 4; ++j) {
    const int f = n0 + 16 * j + nn;
    const float bo = bias[f];
#pragma unroll
    for (int i = 0; i < 4; ++i)
#pragma unroll
      for (int r = 0; r < 8; ++r) {
        const int m = m0 + 16 * i + r + mo;
        out[(size_t)m * DIM_ + f] = acc[i][j][r] + bo;
      }
  }
}

// ---------------------------------------------------------------------------
// Host launcher
// ---------------------------------------------------------------------------
extern "C" void kernel_launch(void* const* d_in, const int* in_sizes, int n_in,
                              void* d_out, int out_size, void* d_ws, size_t ws_size,
                              hipStream_t stream) {
  const float* x = (const float*)d_in[0];
  const unsigned char* mask = (const unsigned char*)d_in[1];
  const float* w_qkv = (const float*)d_in[2];
  const float* w_out = (const float*)d_in[3];
  const float* b_out = (const float*)d_in[4];
  float* out = (float*)d_out;

  const size_t XB_E = (size_t)B_ * N_ * DIM_;   // 4,194,304
  const size_t WQ_E = (size_t)3 * DIM_ * DIM_;  // 3,145,728
  const size_t WO_E = (size_t)DIM_ * DIM_;      // 1,048,576
  const size_t HD_E = (size_t)B_ * H_ * N_ * D_;// 4,194,304
  // total workspace: 48 MB
  char* w = (char*)d_ws;
  bf16* Xb = (bf16*)w; w += XB_E * 2;
  bf16* Wq = (bf16*)w; w += WQ_E * 2;
  bf16* Wo = (bf16*)w; w += WO_E * 2;
  bf16* Qb = (bf16*)w; w += HD_E * 2;
  bf16* Kb = (bf16*)w; w += HD_E * 2;
  bf16* Vt = (bf16*)w; w += HD_E * 2;
  bf16* Ab = (bf16*)w; w += XB_E * 2;
  (void)in_sizes; (void)n_in; (void)out_size; (void)ws_size;

  cvt_f32_bf16<<<(int)(XB_E / 1024), 256, 0, stream>>>(x, Xb, (int)(XB_E / 4));
  cvt_f32_bf16<<<(int)(WQ_E / 1024), 256, 0, stream>>>(w_qkv, Wq, (int)(WQ_E / 4));
  cvt_f32_bf16<<<(int)(WO_E / 1024), 256, 0, stream>>>(w_out, Wo, (int)(WO_E / 4));

  qkv_gemm<<<dim3(32, 12), 256, 0, stream>>>(Xb, Wq, Qb, Kb, Vt);     // 4096x3072
  attn_kernel<<<dim3(B_ * H_, N_ / 64), 128, 0, stream>>>(Qb, Kb, Vt, mask, Ab);
  out_gemm<<<dim3(32, 4), 256, 0, stream>>>(Ab, Wo, b_out, out);      // 4096x1024
}
